// deepSNN_9972914062056
// MI455X (gfx1250) — matrix operand order repaired and checked by reading, not compiled, not run on hardware
//
#include <hip/hip_runtime.h>
#include <hip/hip_bf16.h>

typedef __attribute__((ext_vector_type(16))) __bf16 v16bf;
typedef __attribute__((ext_vector_type(8)))  float  v8f;

// ---------------------------------------------------------------------------
// fp32 -> bf16 elementwise convert
// ---------------------------------------------------------------------------
__global__ __launch_bounds__(256) void f32_to_bf16_kern(const float* __restrict__ in,
                                                        __bf16* __restrict__ out,
                                                        size_t n) {
    size_t i = (size_t)blockIdx.x * 256 + threadIdx.x;
    if (i < n) out[i] = (__bf16)in[i];
}

// ---------------------------------------------------------------------------
// Pack conv weights (Cout,Cin,KH,KW) fp32 into WMMA A-fragment order, bf16.
// Layout: [mCoarse][kchunk][sub(MSUB)][lane(32)][elem(16)]
//   -> each (mCoarse,kchunk) A-tile is MSUB*512 contiguous halfs (async-copy
//      friendly).
// A-matrix 16x32 bf16 per-lane layout used here (contiguous k, B staged in the same order):
//   lane L: hi = L>>4, m = tile*16 + (L&15)
//   elem e: k_local = e + 16*hi
// Zero-pad for m >= Cout or k >= Cin*KH*KW.
// ---------------------------------------------------------------------------
__global__ __launch_bounds__(256) void pack_w_kern(const float* __restrict__ w,
                                                   __bf16* __restrict__ wp,
                                                   int Cout, int Cin, int KH, int KW,
                                                   int mcoarse, int msub, int kch) {
    int idx = blockIdx.x * 256 + threadIdx.x;
    int total = mcoarse * kch * msub * 512;
    if (idx >= total) return;
    int e    = idx & 15;
    int lane = (idx >> 4) & 31;
    int rest = idx >> 9;
    int s    = rest % msub;  rest /= msub;
    int kc   = rest % kch;
    int mtC  = rest / kch;
    int hi   = lane >> 4;
    int m    = (mtC * msub + s) * 16 + (lane & 15);
    int kl   = e + 16 * hi;   // e + 16*hi
    int kg   = kc * 32 + kl;
    int K    = Cin * KH * KW;
    float v  = 0.0f;
    if (m < Cout && kg < K) {
        int ic = kg / (KH * KW);
        int r  = kg % (KH * KW);
        v = w[((m * Cin + ic) * KH + (r / KW)) * KW + (r % KW)];
    }
    wp[idx] = (__bf16)v;
}

// ---------------------------------------------------------------------------
// Implicit-GEMM convolution (pad=PAD) + fire, bf16 WMMA.
//   Block tile: M = MSUB*16 (Cout rows), N = 64 (spatial cols, 16/wave),
//   K = Cin*KH*KW in chunks of 32.
// Block = 128 threads = 4 wave32s. Grid = (ceil(N/64), ceil(Cout/(MSUB*16)), T).
// A tile staged global->LDS via GLOBAL_LOAD_ASYNC_TO_LDS_B128 (ASYNCcnt);
// B tile im2col-gathered by VALU (predicated zero-fill for padding).
// Each wave: 1 B fragment + MSUB A fragments -> MSUB WMMAs per K-chunk.
// !FINAL: writes bf16 spikes (0/1).  FINAL: writes f32 spk and thresholded pot.
// ---------------------------------------------------------------------------
template <int CIN, int KH, int KW, int HIN, int WIN, int PAD,
          int HOUT, int WOUT, int COUT, int MSUB, bool FINAL>
__global__ __launch_bounds__(128)
void conv_wmma_fire(const __bf16* __restrict__ in,     // (T,CIN,HIN,WIN) bf16
                    const __bf16* __restrict__ wpack,  // fragment-packed weights
                    __bf16* __restrict__ spk_bf,       // used when !FINAL
                    float* __restrict__ out_spk,       // used when FINAL
                    float* __restrict__ out_pot,       // used when FINAL
                    float thr) {
    constexpr int K    = CIN * KH * KW;
    constexpr int KCH  = (K + 31) / 32;
    constexpr int NOUT = HOUT * WOUT;
    constexpr int ATILE_HALFS = MSUB * 512;            // A tile per (mCoarse,kc)

    const int ntile = blockIdx.x;
    const int mtile = blockIdx.y;
    const int t     = blockIdx.z;
    const int tid   = threadIdx.x;
    const int wave  = tid >> 5;
    const int lane  = tid & 31;

    __shared__ __align__(16) __bf16 ldsA[ATILE_HALFS]; // [sub][lane][elem]
    __shared__ __align__(16) __bf16 ldsB[64 * 32];     // [n_local][k] row-major

    const __bf16* inT = in + (size_t)t * CIN * HIN * WIN;
    v8f acc[MSUB];
#pragma unroll
    for (int s = 0; s < MSUB; ++s) acc[s] = (v8f){0.f, 0.f, 0.f, 0.f, 0.f, 0.f, 0.f, 0.f};

    const int n_base = ntile * 64;
    // B staging: thread handles n_local = tid>>1, k-half = (tid&1)*16
    const int  bn     = tid >> 1;
    const int  bk0    = (tid & 1) * 16;
    const int  n_g    = n_base + bn;
    const bool nvalid = n_g < NOUT;
    const int  oy     = nvalid ? (n_g / WOUT) : 0;
    const int  ox     = nvalid ? (n_g % WOUT) : 0;

    // LDS byte offset of ldsA (low 32 bits of generic pointer == LDS offset)
    const unsigned ldsA_base = (unsigned)(size_t)(void*)&ldsA[0];

    for (int kc = 0; kc < KCH; ++kc) {
        __syncthreads();
        // ---- stage A tile (MSUB*1024 bytes): async global->LDS, 16B per lane
        {
            const __bf16* src = wpack + (size_t)(mtile * KCH + kc) * ATILE_HALFS;
#pragma unroll
            for (int j = 0; j < MSUB / 2; ++j) {
                unsigned byte_off = (unsigned)(j * 128 + tid) * 16u;
                unsigned lds_addr = ldsA_base + byte_off;
                unsigned long long gaddr =
                    (unsigned long long)(size_t)((const char*)src + byte_off);
                asm volatile("global_load_async_to_lds_b128 %0, %1, off"
                             :: "v"(lds_addr), "v"(gaddr) : "memory");
            }
            if (kc + 1 < KCH)   // prefetch next weight tile (global_prefetch_b8)
                __builtin_prefetch(wpack + (size_t)(mtile * KCH + kc + 1) * ATILE_HALFS, 0, 0);
        }
        // ---- stage B tile (im2col gather): 16 halfs per thread
        {
#pragma unroll
            for (int i = 0; i < 16; ++i) {
                int kk = bk0 + i;
                int kg = kc * 32 + kk;
                __bf16 v = (__bf16)0.0f;
                if (nvalid && kg < K) {
                    int ic = kg / (KH * KW);
                    int r  = kg % (KH * KW);
                    int iy = oy + (r / KW) - PAD;
                    int ix = ox + (r % KW) - PAD;
                    if ((unsigned)iy < (unsigned)HIN && (unsigned)ix < (unsigned)WIN)
                        v = inT[((size_t)ic * HIN + iy) * WIN + ix];
                }
                ldsB[bn * 32 + kk] = v;
            }
        }
        asm volatile("s_wait_asynccnt 0" ::: "memory");   // A tile landed in LDS
        __syncthreads();
        // ---- fragments from LDS (contiguous 32B per lane) + MSUB WMMAs
        const int ncol  = (wave << 4) + (lane & 15);
        const int khalf = (lane >> 4) * 16;
        v16bf bfrag = *(const v16bf*)&ldsB[ncol * 32 + khalf];
#pragma unroll
        for (int s = 0; s < MSUB; ++s) {
            v16bf afrag = *(const v16bf*)&ldsA[s * 512 + lane * 16];
            acc[s] = __builtin_amdgcn_wmma_f32_16x16x32_bf16(
                /*neg_a=*/false, afrag, /*neg_b=*/false, bfrag,
                /*c_mod=*/(short)0, acc[s], /*reuse_a=*/false, /*reuse_b=*/false);
        }
    }

    // ---- epilogue: C/D layout -> (m = (lane>>4)*8 + r, n = lane&15), fused fire
    const int n_out = n_base + (wave << 4) + (lane & 15);
    const int mhi   = (lane >> 4) * 8;
    if (n_out < NOUT) {
#pragma unroll
        for (int s = 0; s < MSUB; ++s) {
#pragma unroll
            for (int r = 0; r < 8; ++r) {
                int oc = (mtile * MSUB + s) * 16 + mhi + r;
                if (oc < COUT) {
                    float  pot  = acc[s][r];
                    bool   fire = pot > thr;
                    size_t o    = ((size_t)t * COUT + oc) * NOUT + n_out;
                    if (FINAL) {
                        out_spk[o] = fire ? 1.0f : 0.0f;
                        out_pot[o] = fire ? pot : 0.0f;
                    } else {
                        spk_bf[o] = fire ? (__bf16)1.0f : (__bf16)0.0f;
                    }
                }
            }
        }
    }
}

// ---------------------------------------------------------------------------
// Max-pool (floor mode / VALID) on bf16 spikes.
// ---------------------------------------------------------------------------
template <int KP, int SP>
__global__ __launch_bounds__(256)
void pool_bf16_kern(const __bf16* __restrict__ in, __bf16* __restrict__ out,
                    int C, int H, int W, int Ho, int Wo, int T) {
    size_t idx   = (size_t)blockIdx.x * 256 + threadIdx.x;
    size_t total = (size_t)T * C * Ho * Wo;
    if (idx >= total) return;
    int    wo = (int)(idx % Wo);
    size_t r  = idx / Wo;
    int    ho = (int)(r % Ho);
    r /= Ho;
    int c = (int)(r % C);
    int t = (int)(r / C);
    const __bf16* p = in + ((size_t)t * C + c) * H * W;
    float m = -3.4e38f;
#pragma unroll
    for (int dy = 0; dy < KP; ++dy)
#pragma unroll
        for (int dx = 0; dx < KP; ++dx) {
            float v = (float)p[(size_t)(ho * SP + dy) * W + (wo * SP + dx)];
            m = fmaxf(m, v);
        }
    out[idx] = (__bf16)m;
}

// ---------------------------------------------------------------------------
// Host-side launcher
// ---------------------------------------------------------------------------
extern "C" void kernel_launch(void* const* d_in, const int* in_sizes, int n_in,
                              void* d_out, int out_size, void* d_ws, size_t ws_size,
                              hipStream_t stream) {
    (void)in_sizes; (void)n_in; (void)out_size; (void)ws_size;
    const float* x  = (const float*)d_in[0];   // (20,6,160,160)
    const float* w1 = (const float*)d_in[1];   // (30,6,4,4)
    const float* w2 = (const float*)d_in[2];   // (250,30,3,3)
    const float* w3 = (const float*)d_in[3];   // (200,250,3,3)
    // d_in[4] = layer_idx (always 3 in setup): final-layer path implemented.

    constexpr int T = 20;
    // layer geometry
    constexpr int H1o = 161, W1o = 161;             // conv1 out
    constexpr int H1p = 80,  W1p = 80;              // pool 2,2
    constexpr int H2o = 82,  W2o = 82;              // conv2 out
    constexpr int H2p = 27,  W2p = 27;              // pool 3,3
    constexpr int H3o = 29,  W3o = 29;              // conv3 out
    // M-blocking: conv1 MSUB=2 (32 rows >= 30); conv2/3 MSUB=4 (64 rows/block)
    constexpr int MS1 = 2, MC1 = 1, KC1 = 3;        // 30 ch, K=96
    constexpr int MS2 = 4, MC2 = 4, KC2 = 9;        // 250 ch, K=270
    constexpr int MS3 = 4, MC3 = 4, KC3 = 71;       // 200 ch, K=2250

    // workspace carve-up (bf16 halfs), 256B aligned
    uint8_t* ws = (uint8_t*)d_ws;
    size_t off = 0;
    auto carve = [&](size_t halfs) {
        size_t o = off;
        off = (off + halfs * sizeof(__bf16) + 255) & ~(size_t)255;
        return (__bf16*)(ws + o);
    };
    __bf16* xb   = carve((size_t)T * 6 * 160 * 160);
    __bf16* w1p  = carve((size_t)MC1 * KC1 * MS1 * 512);
    __bf16* w2p  = carve((size_t)MC2 * KC2 * MS2 * 512);
    __bf16* w3p  = carve((size_t)MC3 * KC3 * MS3 * 512);
    __bf16* spk1 = carve((size_t)T * 30 * H1o * W1o);
    __bf16* p1   = carve((size_t)T * 30 * H1p * W1p);
    __bf16* spk2 = carve((size_t)T * 250 * H2o * W2o);
    __bf16* p2   = carve((size_t)T * 250 * H2p * W2p);

    // 1) convert input to bf16
    {
        size_t n = (size_t)T * 6 * 160 * 160;
        f32_to_bf16_kern<<<(unsigned)((n + 255) / 256), 256, 0, stream>>>(x, xb, n);
    }
    // 2) pack weights into WMMA fragment order (A-tile contiguous per K-chunk)
    pack_w_kern<<<(MC1 * KC1 * MS1 * 512 + 255) / 256, 256, 0, stream>>>(
        w1, w1p, 30, 6, 4, 4, MC1, MS1, KC1);
    pack_w_kern<<<(MC2 * KC2 * MS2 * 512 + 255) / 256, 256, 0, stream>>>(
        w2, w2p, 250, 30, 3, 3, MC2, MS2, KC2);
    pack_w_kern<<<(MC3 * KC3 * MS3 * 512 + 255) / 256, 256, 0, stream>>>(
        w3, w3p, 200, 250, 3, 3, MC3, MS3, KC3);

    // 3) conv1 + fire (thr=1): (T,6,160,160) pad2 -> (T,30,161,161) spikes
    {
        dim3 grid((H1o * W1o + 63) / 64, MC1, T);
        conv_wmma_fire<6, 4, 4, 160, 160, 2, H1o, W1o, 30, MS1, false>
            <<<grid, 128, 0, stream>>>(xb, w1p, spk1, nullptr, nullptr, 1.0f);
    }
    // 4) pool 2x2 s2 -> (T,30,80,80)
    {
        size_t n = (size_t)T * 30 * H1p * W1p;
        pool_bf16_kern<2, 2><<<(unsigned)((n + 255) / 256), 256, 0, stream>>>(
            spk1, p1, 30, H1o, W1o, H1p, W1p, T);
    }
    // 5) conv2 + fire (thr=1): pad2 -> (T,250,82,82) spikes
    {
        dim3 grid((H2o * W2o + 63) / 64, MC2, T);
        conv_wmma_fire<30, 3, 3, H1p, W1p, 2, H2o, W2o, 250, MS2, false>
            <<<grid, 128, 0, stream>>>(p1, w2p, spk2, nullptr, nullptr, 1.0f);
    }
    // 6) pool 3x3 s3 -> (T,250,27,27)
    {
        size_t n = (size_t)T * 250 * H2p * W2p;
        pool_bf16_kern<3, 3><<<(unsigned)((n + 255) / 256), 256, 0, stream>>>(
            spk2, p2, 250, H2o, W2o, H2p, W2p, T);
    }
    // 7) conv3 + fire (thr=25): pad2 -> (T,200,29,29); write spk3 | pot3 to d_out
    {
        float* out_spk = (float*)d_out;
        float* out_pot = out_spk + (size_t)T * 200 * H3o * W3o;
        dim3 grid((H3o * W3o + 63) / 64, MC3, T);
        conv_wmma_fire<250, 3, 3, H2p, W2p, 2, H3o, W3o, 200, MS3, true>
            <<<grid, 128, 0, stream>>>(p2, w3p, nullptr, out_spk, out_pot, 25.0f);
    }
}
